// MHAHead_11905649344630
// MI455X (gfx1250) — matrix operand order repaired and checked by reading, not compiled, not run on hardware
//
#include <hip/hip_runtime.h>

typedef __attribute__((ext_vector_type(16))) __bf16 v16bf;
typedef __attribute__((ext_vector_type(8)))  __bf16 v8bf;
typedef __attribute__((ext_vector_type(8)))  float  v8f;

#define SLEN  4096
#define EDIM  512
#define HDIM  64
#define NBATCH 8
#define NEGBIG (-1e30f)
#define QSCALE 0.04419417382415922f   /* 512^-0.5 folded into Q */

static __device__ __forceinline__ v16bf bf_concat(v8bf lo, v8bf hi) {
  return __builtin_shufflevector(lo, hi, 0,1,2,3,4,5,6,7,8,9,10,11,12,13,14,15);
}
static __device__ __forceinline__ v8f wmma_bf16(v16bf a, v16bf b, v8f c) {
  return __builtin_amdgcn_wmma_f32_16x16x32_bf16(false, a, false, b, (short)0, c,
                                                 false, false);
}

// ---------------------------------------------------------------------------
// Kernel 0: repack Wq/Wk/Wv (f32 [512,64]) into bf16 WMMA B-fragment layout.
// frag id = (mat*4 + ntile)*16 + kstep ; per frag: 32 lanes x 16 bf16.
// B-layout: col n = lane%16 ; elems e -> k = kstep*32 + (lane>=16?8:0) + (e&7) + (e>=8?16:0).
// ---------------------------------------------------------------------------
__global__ void wpack_kernel(const float* __restrict__ Wq,
                             const float* __restrict__ Wk,
                             const float* __restrict__ Wv,
                             __bf16* __restrict__ wfrag) {
  int idx = blockIdx.x * blockDim.x + threadIdx.x;
  if (idx >= 3 * 4 * 16 * 32 * 16) return;
  int e    = idx & 15;
  int lane = (idx >> 4) & 31;
  int frag = idx >> 9;
  int ks   = frag & 15;
  int nt   = (frag >> 4) & 3;
  int mat  = frag >> 6;
  int k = ks * 32 + ((lane >= 16) ? 8 : 0) + (e & 7) + ((e >= 8) ? 16 : 0);
  int n = nt * 16 + (lane & 15);
  const float* W = (mat == 0) ? Wq : ((mat == 1) ? Wk : Wv);
  wfrag[idx] = (__bf16)W[k * HDIM + n];
}

// ---------------------------------------------------------------------------
// Kernel 1: QKV projection. One wave per 16-row tile of x.
// 6-deep software pipeline of W fragments so WMMAs overlap L2 load latency.
// q (pre-scaled), k -> bf16 [B,S,64]; v -> bf16 transposed [B,64,S].
// ---------------------------------------------------------------------------
__global__ void qkv_kernel(const float* __restrict__ x,
                           const __bf16* __restrict__ wfrag,
                           const float* __restrict__ bq,
                           const float* __restrict__ bk,
                           const float* __restrict__ bv,
                           __bf16* __restrict__ qo,
                           __bf16* __restrict__ ko,
                           __bf16* __restrict__ vto) {
  const int lane   = threadIdx.x & 31;
  const int wave   = threadIdx.x >> 5;
  const int lane16 = lane & 15;
  const bool hi    = lane >= 16;
  const int tile   = blockIdx.x * 8 + wave;
  const int b      = tile >> 8;            // 256 tiles per batch
  const int mrow   = (tile & 255) * 16;

  const v8f vzero = {0.f,0.f,0.f,0.f,0.f,0.f,0.f,0.f};
  v8f acc[12];
  #pragma unroll
  for (int i = 0; i < 12; ++i) acc[i] = vzero;

  const float* xrow = x + (size_t)(b * SLEN + mrow + lane16) * EDIM;
  const __bf16* wlane = wfrag + lane * 16;

  for (int ks = 0; ks < 16; ++ks) {
    // A-fragment of x (f32 -> bf16): lane<16 K{0..7,16..23}, lane>=16 +8.
    const int k0 = ks * 32 + (hi ? 8 : 0);
    float4 f0 = *(const float4*)(xrow + k0);
    float4 f1 = *(const float4*)(xrow + k0 + 4);
    float4 f2 = *(const float4*)(xrow + k0 + 16);
    float4 f3 = *(const float4*)(xrow + k0 + 20);

    // batch-issue first 6 W fragments while converting A
    v16bf fb[6];
    #pragma unroll
    for (int i = 0; i < 6; ++i) {
      const __bf16* wp = wlane + (size_t)(i * 16 + ks) * 512;
      fb[i] = bf_concat(*(const v8bf*)wp, *(const v8bf*)(wp + 8));
    }

    v16bf a;
    a[0]=(__bf16)f0.x;  a[1]=(__bf16)f0.y;  a[2]=(__bf16)f0.z;  a[3]=(__bf16)f0.w;
    a[4]=(__bf16)f1.x;  a[5]=(__bf16)f1.y;  a[6]=(__bf16)f1.z;  a[7]=(__bf16)f1.w;
    a[8]=(__bf16)f2.x;  a[9]=(__bf16)f2.y;  a[10]=(__bf16)f2.z; a[11]=(__bf16)f2.w;
    a[12]=(__bf16)f3.x; a[13]=(__bf16)f3.y; a[14]=(__bf16)f3.z; a[15]=(__bf16)f3.w;

    // first 6 WMMAs interleaved with loads of fragments 6..11 (buffer reuse)
    #pragma unroll
    for (int i = 0; i < 6; ++i) {
      acc[i] = wmma_bf16(a, fb[i], acc[i]);
      const __bf16* wp = wlane + (size_t)((i + 6) * 16 + ks) * 512;
      fb[i] = bf_concat(*(const v8bf*)wp, *(const v8bf*)(wp + 8));
    }
    #pragma unroll
    for (int i = 0; i < 6; ++i)
      acc[6 + i] = wmma_bf16(a, fb[i], acc[6 + i]);
  }

  const int rbase = mrow + (hi ? 8 : 0);       // C-layout rows for this lane
  #pragma unroll
  for (int nt = 0; nt < 4; ++nt) {
    const int ncol = nt * 16 + lane16;
    const float biq = bq[ncol], bik = bk[ncol], biv = bv[ncol];
    v8bf vpack;
    #pragma unroll
    for (int vr = 0; vr < 8; ++vr) {
      const size_t row = (size_t)(b * SLEN + rbase + vr);
      qo[row * HDIM + ncol] = (__bf16)((acc[nt][vr]     + biq) * QSCALE);
      ko[row * HDIM + ncol] = (__bf16)( acc[4 + nt][vr] + bik);
      vpack[vr]             = (__bf16)( acc[8 + nt][vr] + biv);
    }
    // V transposed: vt[b][dim][s], 8 contiguous s per lane -> one 16B store.
    *(v8bf*)(vto + ((size_t)b * HDIM + ncol) * SLEN + rbase) = vpack;
  }
}

// ---------------------------------------------------------------------------
// Kernel 2: flash attention. One wave per 16-query tile; 32 keys / iteration.
// Issue order per iteration: mask loads (HBM stream) -> K frags + score WMMA
// -> V frag loads -> softmax (VALU hides V latency) -> P transpose -> PV WMMA.
// ---------------------------------------------------------------------------
__global__ void attn_kernel(const __bf16* __restrict__ q,
                            const __bf16* __restrict__ k,
                            const __bf16* __restrict__ vt,
                            const int* __restrict__ mask,
                            float* __restrict__ out) {
  __shared__ __align__(16) __bf16 pslab[8][512];   // per-wave 16x32 P tile
  const int lane   = threadIdx.x & 31;
  const int wave   = threadIdx.x >> 5;
  const int lane16 = lane & 15;
  const bool hi    = lane >= 16;
  const int khalf  = hi ? 16 : 0;
  const int tile   = blockIdx.x * 8 + wave;
  const int b      = tile >> 8;
  const int qbase  = (tile & 255) * 16;
  __bf16* slab = &pslab[wave][0];

  // Q A-fragments (head dim 64 = two K=32 steps), Q already scaled.
  const __bf16* qrow = q + (size_t)(b * SLEN + qbase + lane16) * HDIM + (hi ? 16 : 0);
  const v16bf aq0 = bf_concat(*(const v8bf*)qrow,        *(const v8bf*)(qrow + 8));
  const v16bf aq1 = bf_concat(*(const v8bf*)(qrow + 32), *(const v8bf*)(qrow + 40));

  const v8f vzero = {0.f,0.f,0.f,0.f,0.f,0.f,0.f,0.f};
  v8f acc[4];
  #pragma unroll
  for (int i = 0; i < 4; ++i) acc[i] = vzero;
  float m[8], l[8];
  #pragma unroll
  for (int i = 0; i < 8; ++i) { m[i] = NEGBIG; l[i] = 0.f; }

  const int* mrow0 = mask + ((size_t)b * SLEN + qbase + (hi ? 8 : 0)) * SLEN + lane16;

  for (int kb = 0; kb < SLEN; kb += 32) {
    // --- 1) mask loads first: this is the 512MB HBM stream ---
    int mv0[8], mv1[8];
    #pragma unroll
    for (int vr = 0; vr < 8; ++vr) {
      const int* mp = mrow0 + (size_t)vr * SLEN + kb;
      mv0[vr] = mp[0];
      mv1[vr] = mp[16];
    }
    // prefetch next mask block while this iteration computes
    if (kb + 32 < SLEN)
      __builtin_prefetch(mrow0 + kb + 32, 0, 1);

    // --- 2) K fragments + score WMMAs ---
    const __bf16* kp0 = k + (size_t)(b * SLEN + kb + lane16) * HDIM;
    const __bf16* kp1 = kp0 + 16 * HDIM;
    v16bf bk00 = bf_concat(*(const v8bf*)(kp0 + khalf),      *(const v8bf*)(kp0 + khalf + 8));
    v16bf bk01 = bf_concat(*(const v8bf*)(kp0 + 32 + khalf), *(const v8bf*)(kp0 + 32 + khalf + 8));
    v16bf bk10 = bf_concat(*(const v8bf*)(kp1 + khalf),      *(const v8bf*)(kp1 + khalf + 8));
    v16bf bk11 = bf_concat(*(const v8bf*)(kp1 + 32 + khalf), *(const v8bf*)(kp1 + 32 + khalf + 8));

    v8f s0 = wmma_bf16(aq0, bk00, vzero); s0 = wmma_bf16(aq1, bk01, s0);
    v8f s1 = wmma_bf16(aq0, bk10, vzero); s1 = wmma_bf16(aq1, bk11, s1);

    // --- 3) V fragments issued now; softmax below hides their latency ---
    v16bf vf[4];
    #pragma unroll
    for (int nt = 0; nt < 4; ++nt) {
      const __bf16* vp = vt + ((size_t)b * HDIM + nt * 16 + lane16) * SLEN + kb + khalf;
      vf[nt] = bf_concat(*(const v8bf*)vp, *(const v8bf*)(vp + 8));
    }

    // --- 4) mask + online softmax (row stats across 16-lane groups) ---
    float rescale[8];
    #pragma unroll
    for (int vr = 0; vr < 8; ++vr) {
      const float sv0 = (mv0[vr] > 0) ? NEGBIG : s0[vr];
      const float sv1 = (mv1[vr] > 0) ? NEGBIG : s1[vr];
      float mx = fmaxf(sv0, sv1);
      mx = fmaxf(mx, __shfl_xor(mx, 1));
      mx = fmaxf(mx, __shfl_xor(mx, 2));
      mx = fmaxf(mx, __shfl_xor(mx, 4));
      mx = fmaxf(mx, __shfl_xor(mx, 8));
      const float mn = fmaxf(m[vr], mx);
      rescale[vr] = __expf(m[vr] - mn);     // finite args; underflows to 0
      m[vr] = mn;
      const float e0 = (sv0 <= -1e29f) ? 0.f : __expf(sv0 - mn);
      const float e1 = (sv1 <= -1e29f) ? 0.f : __expf(sv1 - mn);
      float ps = e0 + e1;
      ps += __shfl_xor(ps, 1);
      ps += __shfl_xor(ps, 2);
      ps += __shfl_xor(ps, 4);
      ps += __shfl_xor(ps, 8);
      l[vr] = l[vr] * rescale[vr] + ps;
      // scatter P (C-layout) into LDS tile [row][col], row stride 32
      slab[(vr + (hi ? 8 : 0)) * 32 + lane16]      = (__bf16)e0;
      slab[(vr + (hi ? 8 : 0)) * 32 + 16 + lane16] = (__bf16)e1;
    }
    #pragma unroll
    for (int nt = 0; nt < 4; ++nt)
      #pragma unroll
      for (int vr = 0; vr < 8; ++vr)
        acc[nt][vr] = acc[nt][vr] * rescale[vr];

    // LDS stores are cross-lane consumed: wait for DS completion (same wave,
    // DS ops are in-order so next iteration's stores can't pass these loads).
    asm volatile("s_wait_dscnt 0" ::: "memory");

    // reload P as an A-fragment (16x32 bf16)
    const __bf16* sl = slab + lane16 * 32 + (hi ? 16 : 0);
    const v16bf pa = bf_concat(*(const v8bf*)sl, *(const v8bf*)(sl + 8));

    // --- 5) O += P * V (V fragments already resident) ---
    #pragma unroll
    for (int nt = 0; nt < 4; ++nt)
      acc[nt] = wmma_bf16(pa, vf[nt], acc[nt]);
  }

  // epilogue: divide by row sums; fully-masked rows (l==0) -> 0
  #pragma unroll
  for (int vr = 0; vr < 8; ++vr) {
    const float inv = (l[vr] > 0.f) ? 1.0f / l[vr] : 0.f;
    const size_t row = (size_t)(b * SLEN + qbase + vr + (hi ? 8 : 0));
    #pragma unroll
    for (int nt = 0; nt < 4; ++nt)
      out[row * HDIM + nt * 16 + lane16] = acc[nt][vr] * inv;
  }
}

extern "C" void kernel_launch(void* const* d_in, const int* in_sizes, int n_in,
                              void* d_out, int out_size, void* d_ws, size_t ws_size,
                              hipStream_t stream) {
  const float* x   = (const float*)d_in[0];
  const int*  mask = (const int*)d_in[1];
  const float* Wq  = (const float*)d_in[2];
  const float* bq  = (const float*)d_in[3];
  const float* Wk  = (const float*)d_in[4];
  const float* bk  = (const float*)d_in[5];
  const float* Wv  = (const float*)d_in[6];
  const float* bv  = (const float*)d_in[7];
  float* out = (float*)d_out;

  char* ws = (char*)d_ws;
  __bf16* wfrag = (__bf16*)ws;                                   // 192 KB
  __bf16* qb    = (__bf16*)(ws + (512u << 10));                  // 4 MB
  __bf16* kbuf  = (__bf16*)(ws + (512u << 10) + (4u << 20));     // 4 MB
  __bf16* vtb   = (__bf16*)(ws + (512u << 10) + (8u << 20));     // 4 MB

  wpack_kernel<<<dim3(384), dim3(256), 0, stream>>>(Wq, Wk, Wv, wfrag);
  qkv_kernel  <<<dim3(256), dim3(256), 0, stream>>>(x, wfrag, bq, bk, bv,
                                                    qb, kbuf, vtb);
  attn_kernel <<<dim3(256), dim3(256), 0, stream>>>(qb, kbuf, vtb, mask, out);
}
